// MAF_39195871544005
// MI455X (gfx1250) — matrix operand order repaired and checked by reading, not compiled, not run on hardware
//
#include <hip/hip_runtime.h>

// ---------------------------------------------------------------------------
// MAF inverse (5 layers x 32 autoregressive steps) for MI455X / gfx1250.
// - v_wmma_f32_16x16x32_f16 for all matmuls (f16 in / f32 accumulate)
// - masked f16 weights staged into LDS via global_load_async_to_lds_b128
//   (ASYNCcnt path), H2 aliases H1 (wave-private rows, in-order LDS)
// - 4 independent WMMA accumulator chains; packed-f16 bias+ReLU epilogue
//   (v_pk_add_f16 / v_pk_max_num_f16) to relieve the VALU/DS pipes
// ---------------------------------------------------------------------------

typedef _Float16 hv16 __attribute__((ext_vector_type(16)));
typedef _Float16 hv8  __attribute__((ext_vector_type(8)));
typedef _Float16 hv2  __attribute__((ext_vector_type(2)));
typedef float    fv8  __attribute__((ext_vector_type(8)));

#define DIMX    32
#define HID     256
#define NLAYERS 5
#define BATCH   4096
#define TB      64          // batch rows per block
#define THREADS 128         // 4 wave32s, 16 rows each
#define WSTEP   90112       // masked-weight halves per layer (8192+65536+16384)
#define WCHUNKS 11264       // 180224 B / 16

// LDS byte offsets (dynamic shared, 236288 B total < 320KB per WGP)
#define oW0   0              // 256x32  f16 = 16384 B
#define oW1   16384          // 256x256 f16 = 131072 B
#define oW2   147456         // 64x256  f16 = 32768 B
#define oH    180224         // 64x256  f16 = 32768 B  (H1, then aliased by H2)
#define oXh   212992         // 64x32   f16 = 4096 B
#define oXf   217088         // 64x32   f32 = 8192 B
#define oZ    225280         // 64x32   f32 = 8192 B
#define oOUT  233472         // 64x2    f32 = 512 B
#define oB1   233984         // 256 f32
#define oB2   235008         // 256 f32
#define oB3   236032         // 64  f32
#define SMEM_BYTES 236288

struct MainArgs {
    const float*     z;
    const float*     b0[NLAYERS];
    const float*     b1[NLAYERS];
    const float*     b2[NLAYERS];
    const float*     ls[NLAYERS];
    const float*     ab[NLAYERS];
    const _Float16*  wq;       // premasked f16 weights, NLAYERS * WSTEP halves
    float*           out;
};

// ---- WMMA helpers ---------------------------------------------------------

__device__ __forceinline__ fv8 wmma32(hv16 a, hv16 b, fv8 c) {
    return __builtin_amdgcn_wmma_f32_16x16x32_f16(
        false, a, false, b, (short)0, c, false, false);
}

__device__ __forceinline__ hv16 combine16(hv8 x, hv8 y) {
    hv16 r;
#pragma unroll
    for (int i = 0; i < 8; ++i) { r[i] = x[i]; r[8 + i] = y[i]; }
    return r;
}

// A-matrix 16x32 f16 fragment (ISA 7.12.2): lo lanes K=0..7 & 16..23,
// hi lanes K=8..15 & 24..31.  p = row base (+ kblock*32 halves).
__device__ __forceinline__ hv16 ldA(const _Float16* p, int hi) {
    const hv8* q = (const hv8*)(p + hi * 8);
    return combine16(q[0], q[2]);          // +0B and +32B chunks
}

// B-matrix 32x16 f16 fragment: lane n holds column n; lo lanes K=0..7 & 16..23,
// hi lanes K=8..15 & 24..31.  p = W[n] row base (+ kblock*32 halves).
__device__ __forceinline__ hv16 ldB(const _Float16* p, int hi) {
    const hv8* q = (const hv8*)(p + hi * 8);
    return combine16(q[0], q[2]);          // +0B and +32B chunks
}

// Packed-f16 epilogue: cvt pairs to .l/.h, then v_pk_add_f16 + v_pk_max_num_f16.
// 16 VALU per 16x16 tile instead of 24, and paired halves share one VGPR so
// the stores go out as ds_store_b16 / ds_store_b16_d16_hi.
__device__ __forceinline__ void emit_tile(_Float16* sH, fv8 acc, hv2 bias2,
                                          int mBase, int hi, int col) {
    const hv2 zero2 = {};
#pragma unroll
    for (int r = 0; r < 8; r += 2) {
        hv2 h;
        h[0] = (_Float16)acc[r];
        h[1] = (_Float16)acc[r + 1];
        h = h + bias2;                               // v_pk_add_f16
        h = __builtin_elementwise_max(h, zero2);     // v_pk_max_num_f16
        sH[(mBase + hi * 8 + r + 0) * HID + col] = h[0];
        sH[(mBase + hi * 8 + r + 1) * HID + col] = h[1];
    }
}

// ---- Prep: bake MADE masks into f16 weights -------------------------------
__global__ __launch_bounds__(256)
void maf_prep(const float* __restrict__ W0, const float* __restrict__ W1,
              const float* __restrict__ W2, _Float16* __restrict__ wq, int layer) {
    const int i = blockIdx.x * 256 + threadIdx.x;
    if (i >= WSTEP) return;
    const bool natural = (layer & 1) == 0;
    _Float16* dst = wq + (size_t)layer * WSTEP;
    if (i < 8192) {                               // W0: [256][32]
        int j = i >> 5, c = i & 31;
        int dO = j % 31 + 1;
        int dI = natural ? (c + 1) : (32 - c);
        dst[i] = (_Float16)((dO >= dI) ? W0[i] : 0.0f);
    } else if (i < 73728) {                       // W1: [256][256]
        int e = i - 8192, j = e >> 8, c = e & 255;
        int dO = j % 31 + 1, dI = c % 31 + 1;
        dst[i] = (_Float16)((dO >= dI) ? W1[e] : 0.0f);
    } else {                                      // W2: [64][256], strict >
        int e = i - 73728, j = e >> 8, c = e & 255;
        int jj = j & 31;
        int dO = natural ? (jj + 1) : (32 - jj);
        int dI = c % 31 + 1;
        dst[i] = (_Float16)((dO > dI) ? W2[e] : 0.0f);
    }
}

// ---- Main kernel ----------------------------------------------------------
__global__ __launch_bounds__(THREADS)
void maf_main(MainArgs p) {
    extern __shared__ __align__(16) char smem[];
    _Float16* sW0 = (_Float16*)(smem + oW0);
    _Float16* sW1 = (_Float16*)(smem + oW1);
    _Float16* sW2 = (_Float16*)(smem + oW2);
    _Float16* sH  = (_Float16*)(smem + oH);     // H1, then overwritten by H2
    _Float16* sXh = (_Float16*)(smem + oXh);
    float*    sXf = (float*)(smem + oXf);
    float*    sZ  = (float*)(smem + oZ);
    float*    sOut= (float*)(smem + oOUT);
    float*    sB1 = (float*)(smem + oB1);
    float*    sB2 = (float*)(smem + oB2);
    float*    sB3 = (float*)(smem + oB3);

    const int tid   = threadIdx.x;
    const int lane  = tid & 31;
    const int wave  = tid >> 5;
    const int lo    = lane & 15;
    const int hi    = lane >> 4;
    const int mBase = wave * 16;          // this wave owns rows [mBase, mBase+16)
    const int gRow0 = blockIdx.x * TB;

    for (int layer = 0; layer < NLAYERS; ++layer) {
        __syncthreads();
        // --- Stage this layer's masked f16 weights (176KB) into LDS via the
        //     CDNA5 async global->LDS path (ASYNCcnt), no VGPR round-trip.
        {
            const _Float16* gsrc = p.wq + (size_t)layer * WSTEP;
            const unsigned ldsBase = (unsigned)(uintptr_t)(smem + oW0);
            for (int i = tid; i < WCHUNKS; i += THREADS) {
                unsigned off = (unsigned)i * 16u;
                unsigned lds = ldsBase + off;
                asm volatile("global_load_async_to_lds_b128 %0, %1, %2"
                             :: "v"(lds), "v"(off), "s"(gsrc) : "memory");
            }
            for (int i = tid; i < HID; i += THREADS) {
                sB1[i] = p.b0[layer][i];
                sB2[i] = p.b1[layer][i];
            }
            for (int i = tid; i < 64; i += THREADS) sB3[i] = p.b2[layer][i];
            // Prefetch next layer's weights into L2 while this layer computes.
            if (layer + 1 < NLAYERS) {
                const char* nxt = (const char*)(p.wq + (size_t)(layer + 1) * WSTEP);
                for (int i = tid; i < 1408; i += THREADS)
                    __builtin_prefetch(nxt + (size_t)i * 128, 0, 0);
            }
        }
        // ActNorm inverse: sZ = x_prev * exp(-log_scale) + bias
        for (int i = tid; i < TB * DIMX; i += THREADS) {
            int c = i & 31;
            float xp = (layer == 0) ? p.z[(size_t)(gRow0 + (i >> 5)) * DIMX + c]
                                    : sXf[i];
            sZ[i] = xp * __expf(-p.ls[layer][c]) + p.ab[layer][c];
        }
        asm volatile("s_wait_asynccnt 0" ::: "memory");
        __syncthreads();
        // x starts at zero each MAF layer (reference scan semantics).
        for (int i = tid; i < TB * DIMX; i += THREADS) {
            sXf[i] = 0.0f;
            sXh[i] = (_Float16)0.0f;
        }
        __syncthreads();

        // Hoist this lane's bias values into registers (packed f16 pairs).
        hv2 rB1[16], rB2[16];
        float rB3[4];
#pragma unroll
        for (int nt = 0; nt < 16; ++nt) {
            _Float16 v1 = (_Float16)sB1[nt * 16 + lo];
            _Float16 v2 = (_Float16)sB2[nt * 16 + lo];
            rB1[nt][0] = v1; rB1[nt][1] = v1;
            rB2[nt][0] = v2; rB2[nt][1] = v2;
        }
#pragma unroll
        for (int ct = 0; ct < 4; ++ct) rB3[ct] = sB3[ct * 16 + lo];

        for (int idx = 0; idx < DIMX; ++idx) {
            // ---- GEMM1: H1 = relu(X @ W0^T + b0), K=32, wave-private rows.
            //      4 independent WMMA chains per group.
            {
                hv16 a = ldA(sXh + (mBase + lo) * DIMX, hi);
#pragma unroll
                for (int nt = 0; nt < 16; nt += 4) {
                    fv8 ac0 = {}, ac1 = {}, ac2 = {}, ac3 = {};
                    hv16 b0 = ldB(sW0 + ((nt + 0) * 16 + lo) * DIMX, hi);
                    hv16 b1 = ldB(sW0 + ((nt + 1) * 16 + lo) * DIMX, hi);
                    hv16 b2 = ldB(sW0 + ((nt + 2) * 16 + lo) * DIMX, hi);
                    hv16 b3 = ldB(sW0 + ((nt + 3) * 16 + lo) * DIMX, hi);
                    ac0 = wmma32(a, b0, ac0);
                    ac1 = wmma32(a, b1, ac1);
                    ac2 = wmma32(a, b2, ac2);
                    ac3 = wmma32(a, b3, ac3);
                    emit_tile(sH, ac0, rB1[nt + 0], mBase, hi, (nt + 0) * 16 + lo);
                    emit_tile(sH, ac1, rB1[nt + 1], mBase, hi, (nt + 1) * 16 + lo);
                    emit_tile(sH, ac2, rB1[nt + 2], mBase, hi, (nt + 2) * 16 + lo);
                    emit_tile(sH, ac3, rB1[nt + 3], mBase, hi, (nt + 3) * 16 + lo);
                }
            }
            // ---- GEMM2: H2 = relu(H1 @ W1^T + b1), K=256.
            //      All H1 A-fragments preloaded, then H2 overwrites in place.
            {
                hv16 afs[8];
#pragma unroll
                for (int kb = 0; kb < 8; ++kb)
                    afs[kb] = ldA(sH + (mBase + lo) * HID + kb * 32, hi);
#pragma unroll
                for (int nt = 0; nt < 16; nt += 4) {
                    fv8 ac0 = {}, ac1 = {}, ac2 = {}, ac3 = {};
#pragma unroll
                    for (int kb = 0; kb < 8; ++kb) {
                        hv16 b0 = ldB(sW1 + ((nt + 0) * 16 + lo) * HID + kb * 32, hi);
                        hv16 b1 = ldB(sW1 + ((nt + 1) * 16 + lo) * HID + kb * 32, hi);
                        hv16 b2 = ldB(sW1 + ((nt + 2) * 16 + lo) * HID + kb * 32, hi);
                        hv16 b3 = ldB(sW1 + ((nt + 3) * 16 + lo) * HID + kb * 32, hi);
                        ac0 = wmma32(afs[kb], b0, ac0);
                        ac1 = wmma32(afs[kb], b1, ac1);
                        ac2 = wmma32(afs[kb], b2, ac2);
                        ac3 = wmma32(afs[kb], b3, ac3);
                    }
                    emit_tile(sH, ac0, rB2[nt + 0], mBase, hi, (nt + 0) * 16 + lo);
                    emit_tile(sH, ac1, rB2[nt + 1], mBase, hi, (nt + 1) * 16 + lo);
                    emit_tile(sH, ac2, rB2[nt + 2], mBase, hi, (nt + 2) * 16 + lo);
                    emit_tile(sH, ac3, rB2[nt + 3], mBase, hi, (nt + 3) * 16 + lo);
                }
            }
            // ---- GEMM3: only the 2 output column-tiles holding cols idx, 32+idx,
            //      run as two interleaved independent chains.
            {
                hv16 afs[8];
#pragma unroll
                for (int kb = 0; kb < 8; ++kb)
                    afs[kb] = ldA(sH + (mBase + lo) * HID + kb * 32, hi);
                const int ct0 = (idx >> 4);
                const int ct1 = 2 + (idx >> 4);
                fv8 ac0 = {}, ac1 = {};
#pragma unroll
                for (int kb = 0; kb < 8; ++kb) {
                    hv16 b0 = ldB(sW2 + (ct0 * 16 + lo) * HID + kb * 32, hi);
                    hv16 b1 = ldB(sW2 + (ct1 * 16 + lo) * HID + kb * 32, hi);
                    ac0 = wmma32(afs[kb], b0, ac0);
                    ac1 = wmma32(afs[kb], b1, ac1);
                }
                if (lo == (idx & 15)) {       // this lane's column == idx (+32)
                    float b0v = (idx >> 4) ? rB3[1] : rB3[0];
                    float b1v = (idx >> 4) ? rB3[3] : rB3[2];
#pragma unroll
                    for (int r = 0; r < 8; ++r) {
                        int m = (mBase + hi * 8 + r) * 2;
                        sOut[m + 0] = ac0[r] + b0v;
                        sOut[m + 1] = ac1[r] + b1v;
                    }
                }
            }
            __syncthreads();
            // x[:, idx] = z[:, idx] * exp(s) + t   (crosses waves -> barriers)
            if (tid < TB) {
                float s  = sOut[tid * 2 + 0];
                float tt = sOut[tid * 2 + 1];
                float xi = sZ[tid * DIMX + idx] * __expf(s) + tt;
                sXf[tid * DIMX + idx] = xi;
                sXh[tid * DIMX + idx] = (_Float16)xi;
            }
            __syncthreads();
        }
    }

    // Epilogue: write final x (f32) to global.
    __syncthreads();
    float4*       out4 = (float4*)p.out;
    const float4* x4   = (const float4*)sXf;
    for (int i = tid; i < TB * 8; i += THREADS) {
        int row = i >> 3, c = i & 7;
        out4[(size_t)(gRow0 + row) * 8 + c] = x4[i];
    }
}

// ---------------------------------------------------------------------------
extern "C" void kernel_launch(void* const* d_in, const int* in_sizes, int n_in,
                              void* d_out, int out_size, void* d_ws, size_t ws_size,
                              hipStream_t stream) {
    (void)in_sizes; (void)n_in; (void)out_size;
    if (ws_size < (size_t)NLAYERS * WSTEP * sizeof(_Float16)) return;
    _Float16* wq = (_Float16*)d_ws;

    // d_in order: z; then per layer {W0,W1,W2,b0,b1,b2}; then per layer {log_scale,bias}
    for (int l = 0; l < NLAYERS; ++l) {
        const float* W0 = (const float*)d_in[1 + 6 * l + 0];
        const float* W1 = (const float*)d_in[1 + 6 * l + 1];
        const float* W2 = (const float*)d_in[1 + 6 * l + 2];
        maf_prep<<<(WSTEP + 255) / 256, 256, 0, stream>>>(W0, W1, W2, wq, l);
    }

    MainArgs a;
    a.z = (const float*)d_in[0];
    for (int l = 0; l < NLAYERS; ++l) {
        a.b0[l] = (const float*)d_in[1 + 6 * l + 3];
        a.b1[l] = (const float*)d_in[1 + 6 * l + 4];
        a.b2[l] = (const float*)d_in[1 + 6 * l + 5];
        a.ls[l] = (const float*)d_in[1 + 6 * NLAYERS + 2 * l + 0];
        a.ab[l] = (const float*)d_in[1 + 6 * NLAYERS + 2 * l + 1];
    }
    a.wq  = wq;
    a.out = (float*)d_out;

    maf_main<<<BATCH / TB, THREADS, SMEM_BYTES, stream>>>(a);
}
